// EGNNBlock_23012434772667
// MI455X (gfx1250) — matrix operand order repaired and checked by reading, not compiled, not run on hardware
//
#include <hip/hip_runtime.h>

// EGNN block for MI455X (gfx1250), wave32 + WMMA bf16.
// d_out = [h_out (50000x128 f32)] ++ [x_out (50000x3 f32)]

#define N_NODES 50000
#define N_EDGES 800000
#define HDIM    128
#define EIN_ACT 329
#define EIN_PAD 352          // 11 K-tiles of 32
#define LDK_E   360          // padded LDS row stride for msg_in tile
#define ROWS_E  32           // edges per wave (2 WMMA M-tiles)

typedef __attribute__((ext_vector_type(16))) __bf16 v16bf;
typedef __attribute__((ext_vector_type(8)))  float  v8f;

union BFrag {
    v16bf v;
    unsigned short s[16];
    uint4 q[2];
};

__device__ __forceinline__ unsigned short f2bf(float f) {
    union { float f; unsigned u; } c; c.f = f;
    unsigned r = c.u + 0x7FFFu + ((c.u >> 16) & 1u);   // round-to-nearest-even
    return (unsigned short)(r >> 16);
}

__device__ __forceinline__ float silu_f(float v) {
    return v * (1.0f / (1.0f + __expf(-v)));
}

// A fragment (16x32 bf16, ISA 7.12.2): lane&15 = M row; lanes 0-15 hold K 0-7 &
// 16-23, lanes 16-31 hold K 8-15 & 24-31.
__device__ __forceinline__ v16bf load_a_frag(const unsigned short* tile, int ldk,
                                             int kbase, int lane) {
    BFrag f;
    const unsigned short* row = tile + (lane & 15) * ldk + kbase + ((lane >> 4) << 3);
#pragma unroll
    for (int i = 0; i < 8; ++i) f.s[i] = row[i];
#pragma unroll
    for (int i = 0; i < 8; ++i) f.s[8 + i] = row[16 + i];
    return f.v;
}

// B fragment by linear fragment index f = kt*8+nt from pre-swizzled weights:
// each lane's 16 bf16 K-values are contiguous (32 bytes) -> two uint4 loads.
__device__ __forceinline__ v16bf load_b_frag_lin(const unsigned short* w, int f,
                                                 int lane) {
    BFrag fr;
    const uint4* p = (const uint4*)(w + (((f * 32 + lane) << 4)));
    fr.q[0] = p[0];
    fr.q[1] = p[1];
    return fr.v;
}

#define WMMA_BF16(A, B, C) \
    __builtin_amdgcn_wmma_f32_16x16x32_bf16(false, (A), false, (B), (short)0, (C), false, false)

// ---------------------------------------------------------------- utility ----

__global__ void zero_kernel(float* p, int n) {
    int i = blockIdx.x * blockDim.x + threadIdx.x;
    if (i < n) p[i] = 0.0f;
}

// Convert fp32 weight [Kact x 128] (row-major, fan_in x fan_out) into bf16
// B-fragment swizzled order, zero-padding K up to Kpad.
__global__ void prep_weight(const float* __restrict__ w, unsigned short* __restrict__ out,
                            int Kact, int Kpad) {
    int idx = blockIdx.x * blockDim.x + threadIdx.x;
    if (idx >= Kpad * HDIM) return;
    int k = idx >> 7;
    int n = idx & 127;
    float v = (k < Kact) ? w[k * HDIM + n] : 0.0f;
    int kt = k >> 5, kr = k & 31;
    int half = (kr >> 3) & 1, i = (kr & 7) + ((kr >> 4) << 3);
    int nt = n >> 4;
    int lane = (n & 15) + (half << 4);
    out[(((kt * 8 + nt) * 32 + lane) << 4) + i] = f2bf(v);
}

// ------------------------------------------------------------- edge kernel ---
// One wave (32 threads) per workgroup; each wave owns 32 edges = two 16-row
// WMMA M-tiles, so every B fragment load feeds two v_wmma ops. B fragments are
// software double-buffered so the WMMAs always run behind an in-flight load.

__global__ __launch_bounds__(32) void edge_kernel(
    const float* __restrict__ h, const float* __restrict__ x,
    const float* __restrict__ edge_attr, const float* __restrict__ t_emb,
    const unsigned short* __restrict__ We1b, const float* __restrict__ We1_bias,
    const unsigned short* __restrict__ We2b, const float* __restrict__ We2_bias,
    const float* __restrict__ Watt_w, const float* __restrict__ Watt_b,
    const unsigned short* __restrict__ Wx1b, const float* __restrict__ Wx1_bias,
    const float* __restrict__ Wx2_w, const float* __restrict__ Wx2_b,
    const int* __restrict__ srcI, const int* __restrict__ dstI,
    float* __restrict__ msg_agg, float* __restrict__ coord_agg, float* __restrict__ deg)
{
    __shared__ unsigned short msgin[ROWS_E][LDK_E];  // 23040 B
    __shared__ unsigned short actt[ROWS_E][HDIM];    //  8192 B
    __shared__ float mtile[ROWS_E][HDIM];            // 16384 B
    __shared__ float diffs[ROWS_E][3];
    __shared__ float scal[ROWS_E];
    __shared__ float dsq[ROWS_E];
    __shared__ int   srcs[ROWS_E];
    __shared__ int   dsts[ROWS_E];

    const int lane = threadIdx.x & 31;
    const int e0 = blockIdx.x * ROWS_E;              // grid exact: E/32 blocks
    const int n0 = lane & 15;
    const int mhi = (lane >> 4) << 3;

    // Per-edge setup: one edge per lane
    {
        int e = e0 + lane;
        int s = srcI[e], d = dstI[e];
        srcs[lane] = s;
        dsts[lane] = d;
        float dx = x[s * 3 + 0] - x[d * 3 + 0];
        float dy = x[s * 3 + 1] - x[d * 3 + 1];
        float dz = x[s * 3 + 2] - x[d * 3 + 2];
        diffs[lane][0] = dx;
        diffs[lane][1] = dy;
        diffs[lane][2] = dz;
        dsq[lane] = dx * dx + dy * dy + dz * dz;
    }
    __syncthreads();

    // Build msg_in tile (32 x 352 bf16): h[src] | h[dst] | dist_sq | edge_attr | t_emb[src] | pad
    for (int row = 0; row < ROWS_E; ++row) {
        const int s = srcs[row];
        const int d = dsts[row];
        const float ds2 = dsq[row];
        const long eidx = (long)(e0 + row);
        for (int c = lane; c < EIN_PAD; c += 32) {
            float v;
            if      (c < 128)  v = h[(long)s * HDIM + c];
            else if (c < 256)  v = h[(long)d * HDIM + (c - 128)];
            else if (c == 256) v = ds2;
            else if (c < 265)  v = edge_attr[eidx * 8 + (c - 257)];
            else if (c < EIN_ACT) v = t_emb[(long)s * 64 + (c - 265)];
            else v = 0.0f;
            msgin[row][c] = f2bf(v);
        }
    }
    __syncthreads();

    v8f acc0[8], acc1[8];
    v8f vzero = {};

    // GEMM1: (32x352) @ We1 (352x128); pipelined B, 2 WMMAs per B fragment
#pragma unroll
    for (int nt = 0; nt < 8; ++nt) { acc0[nt] = vzero; acc1[nt] = vzero; }
    {
        const int FMAX = 11 * 8 - 1;
        v16bf b = load_b_frag_lin(We1b, 0, lane);
        for (int kt = 0; kt < 11; ++kt) {
            v16bf a0 = load_a_frag(&msgin[0][0],  LDK_E, kt * 32, lane);
            v16bf a1 = load_a_frag(&msgin[16][0], LDK_E, kt * 32, lane);
#pragma unroll
            for (int nt = 0; nt < 8; ++nt) {
                int f = kt * 8 + nt;
                v16bf bn = load_b_frag_lin(We1b, (f < FMAX) ? f + 1 : FMAX, lane);
                acc0[nt] = WMMA_BF16(a0, b, acc0[nt]);
                acc1[nt] = WMMA_BF16(a1, b, acc1[nt]);
                b = bn;
            }
        }
    }
#pragma unroll
    for (int nt = 0; nt < 8; ++nt) {
        float bia = We1_bias[nt * 16 + n0];
#pragma unroll
        for (int r = 0; r < 8; ++r) {
            actt[r + mhi][nt * 16 + n0]      = f2bf(silu_f(acc0[nt][r] + bia));
            actt[16 + r + mhi][nt * 16 + n0] = f2bf(silu_f(acc1[nt][r] + bia));
        }
    }
    __syncthreads();

    // GEMM2: (32x128) @ We2 (128x128) -> m (fp32 tile)
#pragma unroll
    for (int nt = 0; nt < 8; ++nt) { acc0[nt] = vzero; acc1[nt] = vzero; }
    {
        const int FMAX = 4 * 8 - 1;
        v16bf b = load_b_frag_lin(We2b, 0, lane);
        for (int kt = 0; kt < 4; ++kt) {
            v16bf a0 = load_a_frag(&actt[0][0],  HDIM, kt * 32, lane);
            v16bf a1 = load_a_frag(&actt[16][0], HDIM, kt * 32, lane);
#pragma unroll
            for (int nt = 0; nt < 8; ++nt) {
                int f = kt * 8 + nt;
                v16bf bn = load_b_frag_lin(We2b, (f < FMAX) ? f + 1 : FMAX, lane);
                acc0[nt] = WMMA_BF16(a0, b, acc0[nt]);
                acc1[nt] = WMMA_BF16(a1, b, acc1[nt]);
                b = bn;
            }
        }
    }
#pragma unroll
    for (int nt = 0; nt < 8; ++nt) {
        float bia = We2_bias[nt * 16 + n0];
#pragma unroll
        for (int r = 0; r < 8; ++r) {
            mtile[r + mhi][nt * 16 + n0]      = acc0[nt][r] + bia;
            mtile[16 + r + mhi][nt * 16 + n0] = acc1[nt][r] + bia;
        }
    }
    __syncthreads();

    // Attention gate: sigmoid(m . Watt + b), one edge per lane (full wave)
    {
        float t = Watt_b[0];
        for (int c = 0; c < HDIM; ++c) t += mtile[lane][c] * Watt_w[c];
        scal[lane] = 1.0f / (1.0f + __expf(-t));
    }
    __syncthreads();

    // m *= att; scatter msg_agg; stage bf16 for coord MLP
    for (int idx = lane; idx < ROWS_E * HDIM; idx += 32) {
        int row = idx >> 7, c = idx & 127;
        float v = mtile[row][c] * scal[row];
        actt[row][c] = f2bf(v);
        unsafeAtomicAdd(&msg_agg[(long)dsts[row] * HDIM + c], v);
    }
    __syncthreads();

    // GEMM3: m_scaled @ Wx1 (128x128) -> silu
#pragma unroll
    for (int nt = 0; nt < 8; ++nt) { acc0[nt] = vzero; acc1[nt] = vzero; }
    {
        const int FMAX = 4 * 8 - 1;
        v16bf b = load_b_frag_lin(Wx1b, 0, lane);
        for (int kt = 0; kt < 4; ++kt) {
            v16bf a0 = load_a_frag(&actt[0][0],  HDIM, kt * 32, lane);
            v16bf a1 = load_a_frag(&actt[16][0], HDIM, kt * 32, lane);
#pragma unroll
            for (int nt = 0; nt < 8; ++nt) {
                int f = kt * 8 + nt;
                v16bf bn = load_b_frag_lin(Wx1b, (f < FMAX) ? f + 1 : FMAX, lane);
                acc0[nt] = WMMA_BF16(a0, b, acc0[nt]);
                acc1[nt] = WMMA_BF16(a1, b, acc1[nt]);
                b = bn;
            }
        }
    }
#pragma unroll
    for (int nt = 0; nt < 8; ++nt) {
        float bia = Wx1_bias[nt * 16 + n0];
#pragma unroll
        for (int r = 0; r < 8; ++r) {
            mtile[r + mhi][nt * 16 + n0]      = silu_f(acc0[nt][r] + bia);
            mtile[16 + r + mhi][nt * 16 + n0] = silu_f(acc1[nt][r] + bia);
        }
    }
    __syncthreads();

    // coord_w = tanh(. @ Wx2 + b); scatter coord_agg & deg (one edge per lane)
    {
        float t = Wx2_b[0];
        for (int c = 0; c < HDIM; ++c) t += mtile[lane][c] * Wx2_w[c];
        float cw = tanhf(t);
        int d = dsts[lane];
        unsafeAtomicAdd(&coord_agg[d * 3 + 0], diffs[lane][0] * cw);
        unsafeAtomicAdd(&coord_agg[d * 3 + 1], diffs[lane][1] * cw);
        unsafeAtomicAdd(&coord_agg[d * 3 + 2], diffs[lane][2] * cw);
        unsafeAtomicAdd(&deg[d], 1.0f);
    }
}

// ------------------------------------------------------------- node kernel ---

#define WAVES_N 4

__global__ __launch_bounds__(128) void node_kernel(
    const float* __restrict__ h, const float* __restrict__ x,
    const float* __restrict__ msg_agg, const float* __restrict__ coord_agg,
    const float* __restrict__ deg,
    const unsigned short* __restrict__ Wh1b, const float* __restrict__ Wh1_bias,
    const unsigned short* __restrict__ Wh2b, const float* __restrict__ Wh2_bias,
    float* __restrict__ h_out, float* __restrict__ x_out)
{
    __shared__ unsigned short hin[WAVES_N][16][256];
    __shared__ unsigned short actt[WAVES_N][16][HDIM];

    const int wave = threadIdx.x >> 5;
    const int lane = threadIdx.x & 31;
    int tile = blockIdx.x * WAVES_N + wave;
    const int ntiles = N_NODES / 16;            // 3125, exact
    if (tile >= ntiles) tile = ntiles - 1;      // duplicate work (same values), uniform barriers
    const int node0 = tile * 16;
    const int n0 = lane & 15;
    const int mhi = (lane >> 4) << 3;

    // h_in tile = [h | msg_agg] (16 x 256 bf16)
    for (int row = 0; row < 16; ++row) {
        const long node = node0 + row;
        for (int c = lane; c < 256; c += 32) {
            float v = (c < 128) ? h[node * HDIM + c] : msg_agg[node * HDIM + (c - 128)];
            hin[wave][row][c] = f2bf(v);
        }
    }
    __syncthreads();

    v8f acc[8];
    v8f vzero = {};

    // GEMM: (16x256) @ Wh1 (256x128) -> silu (pipelined B)
#pragma unroll
    for (int nt = 0; nt < 8; ++nt) acc[nt] = vzero;
    {
        const int FMAX = 8 * 8 - 1;
        v16bf b = load_b_frag_lin(Wh1b, 0, lane);
        for (int kt = 0; kt < 8; ++kt) {
            v16bf a = load_a_frag(&hin[wave][0][0], 256, kt * 32, lane);
#pragma unroll
            for (int nt = 0; nt < 8; ++nt) {
                int f = kt * 8 + nt;
                v16bf bn = load_b_frag_lin(Wh1b, (f < FMAX) ? f + 1 : FMAX, lane);
                acc[nt] = WMMA_BF16(a, b, acc[nt]);
                b = bn;
            }
        }
    }
#pragma unroll
    for (int nt = 0; nt < 8; ++nt) {
        float bia = Wh1_bias[nt * 16 + n0];
#pragma unroll
        for (int r = 0; r < 8; ++r)
            actt[wave][r + mhi][nt * 16 + n0] = f2bf(silu_f(acc[nt][r] + bia));
    }
    __syncthreads();

    // GEMM: (16x128) @ Wh2 (128x128) + bias + residual (pipelined B)
#pragma unroll
    for (int nt = 0; nt < 8; ++nt) acc[nt] = vzero;
    {
        const int FMAX = 4 * 8 - 1;
        v16bf b = load_b_frag_lin(Wh2b, 0, lane);
        for (int kt = 0; kt < 4; ++kt) {
            v16bf a = load_a_frag(&actt[wave][0][0], HDIM, kt * 32, lane);
#pragma unroll
            for (int nt = 0; nt < 8; ++nt) {
                int f = kt * 8 + nt;
                v16bf bn = load_b_frag_lin(Wh2b, (f < FMAX) ? f + 1 : FMAX, lane);
                acc[nt] = WMMA_BF16(a, b, acc[nt]);
                b = bn;
            }
        }
    }
#pragma unroll
    for (int nt = 0; nt < 8; ++nt) {
        float bia = Wh2_bias[nt * 16 + n0];
#pragma unroll
        for (int r = 0; r < 8; ++r) {
            long o = (long)(node0 + r + mhi) * HDIM + nt * 16 + n0;
            h_out[o] = h[o] + acc[nt][r] + bia;
        }
    }

    // x_out = x + coord_agg / (deg + 1)
    if (lane < 16) {
        long node = node0 + lane;
        float dg = deg[node] + 1.0f;
#pragma unroll
        for (int j = 0; j < 3; ++j)
            x_out[node * 3 + j] = x[node * 3 + j] + coord_agg[node * 3 + j] / dg;
    }
}

// ----------------------------------------------------------------- launch ----

extern "C" void kernel_launch(void* const* d_in, const int* in_sizes, int n_in,
                              void* d_out, int out_size, void* d_ws, size_t ws_size,
                              hipStream_t stream) {
    (void)in_sizes; (void)n_in; (void)out_size; (void)ws_size;

    const float* h         = (const float*)d_in[0];
    const float* x         = (const float*)d_in[1];
    const float* edge_attr = (const float*)d_in[2];
    const float* t_emb     = (const float*)d_in[3];
    const float* We1_w  = (const float*)d_in[4];
    const float* We1_b  = (const float*)d_in[5];
    const float* We2_w  = (const float*)d_in[6];
    const float* We2_b  = (const float*)d_in[7];
    const float* Watt_w = (const float*)d_in[8];
    const float* Watt_b = (const float*)d_in[9];
    const float* Wx1_w  = (const float*)d_in[10];
    const float* Wx1_b  = (const float*)d_in[11];
    const float* Wx2_w  = (const float*)d_in[12];
    const float* Wx2_b  = (const float*)d_in[13];
    const float* Wh1_w  = (const float*)d_in[14];
    const float* Wh1_b  = (const float*)d_in[15];
    const float* Wh2_w  = (const float*)d_in[16];
    const float* Wh2_b  = (const float*)d_in[17];
    const int* edge_index = (const int*)d_in[18];
    const int* srcI = edge_index;            // edge_index[0, :]
    const int* dstI = edge_index + N_EDGES;  // edge_index[1, :]

    // Workspace layout: bf16 swizzled weights, then fp32 accumulators.
    unsigned short* We1b = (unsigned short*)d_ws;
    unsigned short* We2b = We1b + EIN_PAD * HDIM;
    unsigned short* Wx1b = We2b + HDIM * HDIM;
    unsigned short* Wh1b = Wx1b + HDIM * HDIM;
    unsigned short* Wh2b = Wh1b + 256 * HDIM;
    float* msg_agg   = (float*)(Wh2b + HDIM * HDIM);         // N x 128
    float* coord_agg = msg_agg + (size_t)N_NODES * HDIM;     // N x 3
    float* deg       = coord_agg + (size_t)N_NODES * 3;      // N

    float* h_out = (float*)d_out;
    float* x_out = h_out + (size_t)N_NODES * HDIM;

    // Zero accumulators (contiguous: 132 floats per node)
    int zn = N_NODES * (HDIM + 3 + 1);
    zero_kernel<<<(zn + 255) / 256, 256, 0, stream>>>(msg_agg, zn);

    // Weight preconversion to bf16 fragment order
    prep_weight<<<(EIN_PAD * HDIM + 255) / 256, 256, 0, stream>>>(We1_w, We1b, EIN_ACT, EIN_PAD);
    prep_weight<<<(HDIM * HDIM + 255) / 256, 256, 0, stream>>>(We2_w, We2b, HDIM, HDIM);
    prep_weight<<<(HDIM * HDIM + 255) / 256, 256, 0, stream>>>(Wx1_w, Wx1b, HDIM, HDIM);
    prep_weight<<<(256 * HDIM + 255) / 256, 256, 0, stream>>>(Wh1_w, Wh1b, 256, 256);
    prep_weight<<<(HDIM * HDIM + 255) / 256, 256, 0, stream>>>(Wh2_w, Wh2b, HDIM, HDIM);

    // Edge pipeline: one wave per block, 32 edges per wave (two M-tiles)
    edge_kernel<<<N_EDGES / ROWS_E, 32, 0, stream>>>(
        h, x, edge_attr, t_emb,
        We1b, We1_b, We2b, We2_b, Watt_w, Watt_b, Wx1b, Wx1_b, Wx2_w, Wx2_b,
        srcI, dstI, msg_agg, coord_agg, deg);

    // Node pipeline: 3125 tiles of 16 nodes, 4 tiles per 128-thread block
    node_kernel<<<((N_NODES / 16) + WAVES_N - 1) / WAVES_N, 128, 0, stream>>>(
        h, x, msg_agg, coord_agg, deg,
        Wh1b, Wh1_b, Wh2b, Wh2_b, h_out, x_out);
}
